// GCRBi2all_58789512348203
// MI455X (gfx1250) — matrix operand order repaired and checked by reading, not compiled, not run on hardware
//
#include <hip/hip_runtime.h>
#include <hip/hip_bf16.h>

// ---------------------------------------------------------------------------
// Problem constants (match reference)
// ---------------------------------------------------------------------------
#define B_SZ    4096
#define N0      10
#define N1      5
#define NFEAT   500
#define TDIM    128
#define NCLASS  40

#define BLK_M   128      // rows per 256-thread workgroup (16 per wave)
#define KSTEPS  16       // 500 padded to 512 = 16 x 32

typedef __attribute__((ext_vector_type(16))) __bf16 v16bf;
typedef __attribute__((ext_vector_type(8)))  float  v8f;

union Frag32B { uint4 q[2]; v16bf v; };

// native f32->bf16 (backend emits v_cvt_pk_bf16_f32 class ops, RNE)
__device__ __forceinline__ unsigned pack2(float a, float b) {
    union { __bf16 h[2]; unsigned u; } r;
    r.h[0] = (__bf16)a;
    r.h[1] = (__bf16)b;
    return r.u;
}

// ---------------------------------------------------------------------------
// Setup: pre-swizzle W[500x128] fp32 -> bf16 in WMMA B-fragment order.
// Layout: Wbf[ks*4096 + col*32 + kl], kl = K within the 32-step (zero padded).
// A B-fragment for (ks, ntile, lane) is then two 16-byte runs (kl = 8*half.. and 16 + 8*half..).
// Runs once; 256 KB of strided reads, L2-resident afterwards.
// ---------------------------------------------------------------------------
__global__ __launch_bounds__(256)
void swizzle_w_kernel(const float* __restrict__ W, unsigned short* __restrict__ Wbf)
{
    int idx = blockIdx.x * 256 + threadIdx.x;      // 0 .. 65535
    int ks  = idx >> 12;
    int rem = idx & 4095;
    int col = rem >> 5;
    int kl  = rem & 31;
    int k   = ks * 32 + kl;
    float v = (k < NFEAT) ? W[(size_t)k * TDIM + col] : 0.f;
    union { __bf16 h; unsigned short u; } cv;
    cv.h = (__bf16)v;
    Wbf[idx] = cv.u;
}

// ---------------------------------------------------------------------------
// Kernel 1: fused  E = X[M,500] @ W[500,128]  ->  mean-pool rows by GROUP.
// No LDS, no barriers. Each wave32 owns a 16-row M-tile x full N=128:
//   - A fragments: per-lane direct global loads (4x b128) of its own row,
//     converted to bf16 in registers (lane pair i / i+16 consumes each
//     64B line fully -> no wasted HBM bytes).
//   - B fragments: 2x b128 global loads from the pre-swizzled L2-resident Wbf.
//   - 8x v_wmma_f32_16x16x32_bf16 back-to-back per K-step, fp32 accumulate.
// Epilogue fuses neighbor mean-pooling (GROUP=1: plain store, else f32 atomics).
// ---------------------------------------------------------------------------
template<int GROUP>
__global__ __launch_bounds__(256)
void gemm_pool_kernel(const float* __restrict__ X,
                      const unsigned short* __restrict__ Wbf,
                      float* __restrict__ outp)
{
    const int lane = threadIdx.x & 31;
    const int wave = threadIdx.x >> 5;
    const int row  = lane & 15;
    const int half = lane >> 4;                       // 0 or 1

    const long rowBase = (long)blockIdx.x * BLK_M;
    const float* xr = X + (rowBase + wave * 16 + row) * (long)NFEAT;

    v8f acc[8] = {};
    float4 f[4];

    // A fragment K offsets for this lane: chunk0 = kk + half*8 (+0,+4),
    // chunk1 = kk + 16 + half*8 (+0,+4)   (ISA 7.12.2 16-bit A 16x32 layout)
    auto loadA_fast = [&](int ks) {
        const int b0 = ks * 32 + half * 8;
        f[0] = *(const float4*)(xr + b0);
        f[1] = *(const float4*)(xr + b0 + 4);
        f[2] = *(const float4*)(xr + b0 + 16);
        f[3] = *(const float4*)(xr + b0 + 20);
    };
    auto loadA_tail = [&](int ks) {                   // ks==15: K 480..511, valid < 500
        const int b0 = ks * 32 + half * 8;
        const float4 z = make_float4(0.f, 0.f, 0.f, 0.f);
        f[0] = (b0      < NFEAT) ? *(const float4*)(xr + b0)      : z;
        f[1] = (b0 + 4  < NFEAT) ? *(const float4*)(xr + b0 + 4)  : z;
        f[2] = (b0 + 16 < NFEAT) ? *(const float4*)(xr + b0 + 16) : z;
        f[3] = (b0 + 20 < NFEAT) ? *(const float4*)(xr + b0 + 20) : z;
    };
    auto cvtA = [&]() {
        Frag32B fa;
        unsigned* u = (unsigned*)&fa;
        u[0] = pack2(f[0].x, f[0].y); u[1] = pack2(f[0].z, f[0].w);
        u[2] = pack2(f[1].x, f[1].y); u[3] = pack2(f[1].z, f[1].w);
        u[4] = pack2(f[2].x, f[2].y); u[5] = pack2(f[2].z, f[2].w);
        u[6] = pack2(f[3].x, f[3].y); u[7] = pack2(f[3].z, f[3].w);
        return fa;
    };

    // B fragment base for this lane (col = ntile*16 + row, K = half*8 +0..7 and 16+half*8 +0..7)
    const unsigned short* wbase = Wbf + (size_t)row * 32 + ((size_t)half << 3);

    loadA_fast(0);

    for (int ks = 0; ks < KSTEPS; ++ks) {
        Frag32B fa = cvtA();                           // consume f[] (waits loads)
        if (ks < KSTEPS - 2)      loadA_fast(ks + 1);  // prefetch next A tile
        else if (ks == KSTEPS - 2) loadA_tail(ks + 1);

        // issue all 8 B-fragment loads (L2 hits), then WMMA chain back-to-back
        const uint4* wq = (const uint4*)(wbase + ((size_t)ks << 12));
        Frag32B fb[8];
#pragma unroll
        for (int n = 0; n < 8; ++n) {
            fb[n].q[0] = wq[n * 64];
            fb[n].q[1] = wq[n * 64 + 2];
        }
#pragma unroll
        for (int n = 0; n < 8; ++n) {
            acc[n] = __builtin_amdgcn_wmma_f32_16x16x32_bf16(
                false, fa.v, false, fb[n].v, (short)0, acc[n], false, false);
        }
    }

    // ---- epilogue: fused neighbor mean-pooling ----
    // C/D layout: VGPR i, lane L -> M = i + 8*(L>>4), N = L&15
    const long mBase = rowBase + wave * 16 + (half << 3);
    const int  c0    = lane & 15;
    if (GROUP == 1) {
#pragma unroll
        for (int n = 0; n < 8; ++n)
#pragma unroll
            for (int i = 0; i < 8; ++i)
                outp[(mBase + i) * TDIM + n * 16 + c0] = acc[n][i];
    } else {
        constexpr float inv = 1.0f / (float)GROUP;
#pragma unroll
        for (int n = 0; n < 8; ++n)
#pragma unroll
            for (int i = 0; i < 8; ++i) {
                long R = mBase + i;                    // constant divisor -> mul/shift
                atomicAdd(&outp[(R / GROUP) * TDIM + n * 16 + c0], acc[n][i] * inv);
            }
    }
}

// ---------------------------------------------------------------------------
// Kernel 2: b1 = table[xbi1].reshape(B, N0, T).mean(1)
// ---------------------------------------------------------------------------
__global__ __launch_bounds__(128)
void table_gather_mean(const int* __restrict__ idx, const float* __restrict__ table,
                       float* __restrict__ b1)
{
    const int b = blockIdx.x, t = threadIdx.x;
    float acc = 0.f;
#pragma unroll
    for (int j = 0; j < N0; ++j) {
        int r = idx[b * N0 + j];
        acc += table[(long)r * TDIM + t];
    }
    b1[b * TDIM + t] = acc * (1.f / N0);
}

// ---------------------------------------------------------------------------
// Kernel 3: 12 bilinear terms -> attention pooling -> [128x40] head -> log_softmax
// ---------------------------------------------------------------------------
__global__ __launch_bounds__(128)
void attn_out_kernel(const float* __restrict__ e0, const float* __restrict__ e1,
                     const float* __restrict__ e2, const float* __restrict__ b1,
                     const float* __restrict__ b2,
                     const float* __restrict__ w1, const float* __restrict__ b1s,
                     const float* __restrict__ w2, const float* __restrict__ b2s,
                     float* __restrict__ out)
{
    __shared__ float sc[12 * TDIM];
    __shared__ float score[12];
    __shared__ float hsh[TDIM];
    __shared__ float ov[NCLASS];

    const int b = blockIdx.x, t = threadIdx.x;
    const float x0 = e0[b * TDIM + t];
    const float x1 = e1[b * TDIM + t];
    const float x2 = e2[b * TDIM + t];
    const float y1 = b1[b * TDIM + t];
    const float y2 = b2[b * TDIM + t];

    float c[12] = { x0 * y1, x0 * y2, x1 * y1, x1 * y2, x2 * y1, x2 * y2,
                    x0 * x1, x0 * x2, x1 * x2, x0, x1, x2 };

    const float wv = w1[t];
#pragma unroll
    for (int k = 0; k < 12; ++k) sc[k * TDIM + t] = c[k] * wv;
    __syncthreads();

    if (t < 12) {
        float s = 0.f;
        for (int i = 0; i < TDIM; ++i) s += sc[t * TDIM + i];
        score[t] = s + b1s[0];
    }
    __syncthreads();

    float m = -3.402823e38f;
#pragma unroll
    for (int k = 0; k < 12; ++k) m = fmaxf(m, score[k]);
    float att[12], den = 0.f;
#pragma unroll
    for (int k = 0; k < 12; ++k) { att[k] = __expf(score[k] - m); den += att[k]; }
    float hid = 0.f;
#pragma unroll
    for (int k = 0; k < 12; ++k) hid += att[k] * c[k];
    hid /= den;

    hsh[t] = hid;
    __syncthreads();

    if (t < NCLASS) {
        float s = b2s[t];
        for (int i = 0; i < TDIM; ++i) s += hsh[i] * w2[i * NCLASS + t];
        ov[t] = s;
    }
    __syncthreads();

    if (t < NCLASS) {
        float mm = ov[0];
        for (int j = 1; j < NCLASS; ++j) mm = fmaxf(mm, ov[j]);
        float se = 0.f;
        for (int j = 0; j < NCLASS; ++j) se += __expf(ov[j] - mm);
        out[b * NCLASS + t] = ov[t] - mm - __logf(se);
    }
}

// ---------------------------------------------------------------------------
// Launch
// ---------------------------------------------------------------------------
extern "C" void kernel_launch(void* const* d_in, const int* in_sizes, int n_in,
                              void* d_out, int out_size, void* d_ws, size_t ws_size,
                              hipStream_t stream)
{
    (void)in_sizes; (void)n_in; (void)out_size; (void)ws_size;

    const float* x0    = (const float*)d_in[0];
    const float* x1    = (const float*)d_in[1];
    const float* x2    = (const float*)d_in[2];
    /* d_in[3] = xbi0: computed then discarded in reference -> skipped */
    const int*   xbi1  = (const int*)  d_in[4];
    const float* xbi2  = (const float*)d_in[5];
    const float* Wt    = (const float*)d_in[6];
    const float* table = (const float*)d_in[7];
    const float* l1w   = (const float*)d_in[8];
    const float* l1b   = (const float*)d_in[9];
    const float* l2w   = (const float*)d_in[10];
    const float* l2b   = (const float*)d_in[11];
    float*       out   = (float*)d_out;

    const size_t P = (size_t)B_SZ * TDIM;           // pooled buffer elements
    float* e0 = (float*)d_ws;                       // plain-stored
    float* b1 = e0 + P;                             // plain-stored
    float* e1 = b1 + P;                             // atomic targets (zeroed)
    float* e2 = e1 + P;
    float* b2 = e2 + P;
    unsigned short* Wbf = (unsigned short*)(b2 + P);  // 512*128 bf16 pre-swizzled W

    swizzle_w_kernel<<<256, 256, 0, stream>>>(Wt, Wbf);
    hipMemsetAsync(e1, 0, 3 * P * sizeof(float), stream);

    // Fused GEMM + mean-pool (all row counts % 128 == 0)
    gemm_pool_kernel<1> <<<B_SZ / BLK_M,           256, 0, stream>>>(x0,   Wbf, e0);
    gemm_pool_kernel<N0><<<B_SZ * N0 / BLK_M,      256, 0, stream>>>(x1,   Wbf, e1);
    gemm_pool_kernel<N0 * N1><<<B_SZ * N0 * N1 / BLK_M, 256, 0, stream>>>(x2,   Wbf, e2);
    gemm_pool_kernel<N0 * N1><<<B_SZ * N0 * N1 / BLK_M, 256, 0, stream>>>(xbi2, Wbf, b2);

    table_gather_mean<<<B_SZ, 128, 0, stream>>>(xbi1, table, b1);

    attn_out_kernel<<<B_SZ, 128, 0, stream>>>(e0, e1, e2, b1, b2,
                                              l1w, l1b, l2w, l2b, out);
}
